// LAGCLEncoder_9904194585124
// MI455X (gfx1250) — matrix operand order repaired and checked by reading, not compiled, not run on hardware
//
#include <hip/hip_runtime.h>
#include <hip/hip_bf16.h>

#define DHID 128
#define TAILK 5

typedef __bf16 bf16_t;
typedef __attribute__((ext_vector_type(16))) bf16_t v16bf;
typedef __attribute__((ext_vector_type(8)))  float  v8f;

static __device__ __forceinline__ bf16_t f2bf(float f) {
  unsigned u = __float_as_uint(f);
  u += 0x7FFFu + ((u >> 16) & 1u);          // round-to-nearest-even
  unsigned short s = (unsigned short)(u >> 16);
  return __builtin_bit_cast(bf16_t, s);
}

// ---------------------------------------------------------------- utilities
__global__ void k_zero_i32(int* __restrict__ p, int n) {
  int i = blockIdx.x * blockDim.x + threadIdx.x;
  if (i < n) p[i] = 0;
}

__global__ void k_deg(const int* __restrict__ src, int* __restrict__ deg, int E) {
  int e = blockIdx.x * blockDim.x + threadIdx.x;
  if (e < E) atomicAdd(&deg[src[e]], 1);
}

// block-local inclusive scan -> exclusive per-element, block totals out
__global__ void k_scan_block(const int* __restrict__ deg, int* __restrict__ rs,
                             int* __restrict__ bsum, int n) {
  __shared__ int sm[256];
  const int t = threadIdx.x;
  const int i = blockIdx.x * 256 + t;
  int v = (i < n) ? deg[i] : 0;
  sm[t] = v; __syncthreads();
  for (int off = 1; off < 256; off <<= 1) {
    int add = (t >= off) ? sm[t - off] : 0;
    __syncthreads();
    sm[t] += add;
    __syncthreads();
  }
  if (i < n) rs[i] = sm[t] - v;              // exclusive within block
  if (t == 255) bsum[blockIdx.x] = sm[t];    // block total
}

__global__ void k_scan_bsums(int* __restrict__ bsum, int nb) {
  __shared__ int sm[1024];
  const int t = threadIdx.x;
  int v = (t < nb) ? bsum[t] : 0;
  sm[t] = v; __syncthreads();
  for (int off = 1; off < 1024; off <<= 1) {
    int add = (t >= off) ? sm[t - off] : 0;
    __syncthreads();
    sm[t] += add;
    __syncthreads();
  }
  if (t < nb) bsum[t] = sm[t] - v;           // exclusive block offsets
}

__global__ void k_scan_add(int* __restrict__ rs, const int* __restrict__ bsum,
                           const int* __restrict__ deg, int* __restrict__ rc,
                           int n, int E) {
  int i = blockIdx.x * blockDim.x + threadIdx.x;
  if (i < n) {
    int v = rs[i] + bsum[i >> 8];
    rs[i] = v;
    rc[i] = v;                               // CSR fill cursor
    if (i == n - 1) rs[n] = E;
  }
}

__global__ void k_node_params(const int* __restrict__ deg, float* __restrict__ inv_d,
                              float* __restrict__ inv_d1, float* __restrict__ inv_d2, int n) {
  int i = blockIdx.x * blockDim.x + threadIdx.x;
  if (i < n) {
    float d = (float)deg[i];
    inv_d[i]  = (deg[i] > 0) ? 1.0f / d : 0.0f;
    inv_d1[i] = 1.0f / (d + 1.0f);
    inv_d2[i] = 1.0f / (d + 2.0f);
  }
}

__global__ void k_csr_fill(const int* __restrict__ src, const int* __restrict__ dst,
                           int* __restrict__ rc, int* __restrict__ csr, int E) {
  int e = blockIdx.x * blockDim.x + threadIdx.x;
  if (e < E) {
    int pos = atomicAdd(&rc[src[e]], 1);
    csr[pos] = dst[e];
  }
}

__global__ void k_init(const float* __restrict__ x, float* __restrict__ h,
                       float* __restrict__ emb, int total) {
  int i = blockIdx.x * blockDim.x + threadIdx.x;
  if (i < total) { float v = x[i]; h[i] = v; emb[i] = v; }
}

// --------------------------------------------------- CSR row-sum (wave/row)
// head==1: out = h_next = inv_d1*(S+h), emb += h_next.   head==0: out = S.
__global__ __launch_bounds__(256) void k_rowsum(
    const float* __restrict__ h, const int* __restrict__ rs,
    const int* __restrict__ csr, float* __restrict__ out,
    float* __restrict__ emb, const float* __restrict__ inv_d1,
    int N, int head) {
  const int lane = threadIdx.x & 31;
  const int row  = (blockIdx.x * blockDim.x + threadIdx.x) >> 5;
  if (row >= N) return;
  const int e0 = rs[row], e1 = rs[row + 1];
  float ax = 0.f, ay = 0.f, az = 0.f, aw = 0.f;
  for (int e = e0; e < e1; ++e) {
    const int d = csr[e];
    if (e + 1 < e1) __builtin_prefetch(h + (size_t)csr[e + 1] * DHID, 0, 0);
    const float4 v = ((const float4*)(h + (size_t)d * DHID))[lane];
    ax += v.x; ay += v.y; az += v.z; aw += v.w;
  }
  float4* orow = (float4*)(out + (size_t)row * DHID);
  if (head) {
    const float s = inv_d1[row];
    const float4 hv = ((const float4*)(h + (size_t)row * DHID))[lane];
    float4 hn; hn.x = s * (ax + hv.x); hn.y = s * (ay + hv.y);
    hn.z = s * (az + hv.z); hn.w = s * (aw + hv.w);
    orow[lane] = hn;
    float4* ep = (float4*)(emb + (size_t)row * DHID);
    float4 ev = ep[lane];
    ev.x += hn.x; ev.y += hn.y; ev.z += hn.z; ev.w += hn.w;
    ep[lane] = ev;
  } else {
    float4 sv; sv.x = ax; sv.y = ay; sv.z = az; sv.w = aw;
    orow[lane] = sv;
  }
}

// -------------------------------- pack W = [[G1;G2]|[B1;B2]] col-major bf16
__global__ void k_pack_w(const float* __restrict__ G1, const float* __restrict__ G2,
                         const float* __restrict__ B1, const float* __restrict__ B2,
                         int l, bf16_t* __restrict__ Wt) {
  const int c = blockIdx.x;      // output column 0..255 (gamma | beta)
  const int k = threadIdx.x;     // reduction index 0..255 (h | nb)
  const size_t base = (size_t)l * DHID * DHID;
  float v;
  if (c < DHID)
    v = (k < DHID) ? G1[base + (size_t)c * DHID + k]
                   : G2[base + (size_t)c * DHID + (k - DHID)];
  else
    v = (k < DHID) ? B1[base + (size_t)(c - DHID) * DHID + k]
                   : B2[base + (size_t)(c - DHID) * DHID + (k - DHID)];
  Wt[(size_t)c * 256 + k] = f2bf(v);
}

// ------------------------------------ fused tail hop: WMMA GEMM + epilogue
// graw|braw = [h | inv_d*S] @ Wt^T ; m = h + (lrelu(g)+1)*R + lrelu(b) - nb ;
// h_next = inv_d2*(S + h + m) ; emb += h_next
__global__ __launch_bounds__(256) void k_tail_gemm(
    const float* __restrict__ h, const float* __restrict__ S,
    const bf16_t* __restrict__ Wt, const float* __restrict__ Rl,
    const float* __restrict__ inv_d, const float* __restrict__ inv_d2,
    float* __restrict__ h_next, float* __restrict__ emb,
    int N, int nMB) {
  __shared__ __align__(16) unsigned char smem[33536];
  bf16_t* Xs = (bf16_t*)smem;   // [32][264] bf16, +8 pad -> conflict-free b128 A loads
  float*  As = (float*)smem;    // [32][260] f32 (aliased, after barrier)

  const int t    = threadIdx.x;
  const int wv   = t >> 5;      // wave owns output cols [wv*32, wv*32+32)
  const int lane = t & 31;
  const int half = lane >> 4;
  const int lm   = lane & 15;

  // persistent B fragments: lane lm = col, 16 contiguous K per half-wave (32B)
  v16bf Bf[2][8];
#pragma unroll
  for (int ct = 0; ct < 2; ++ct) {
    const bf16_t* bp = Wt + (size_t)(wv * 32 + ct * 16 + lm) * 256 + half * 16;
#pragma unroll
    for (int k = 0; k < 8; ++k) {
      union { v16bf v; int4 q[2]; } u;
      u.q[0] = *(const int4*)(bp + k * 32);
      u.q[1] = *(const int4*)(bp + k * 32 + 8);
      Bf[ct][k] = u.v;
    }
  }

  for (int mb = blockIdx.x; mb < nMB; mb += gridDim.x) {
    const int r0 = mb * 32;
    __syncthreads();            // smem reuse from previous iteration
    {                           // stage X tile (32 rows x 256 bf16)
      const int row = t >> 3;
      const int seg = t & 7;
      int r = r0 + row; if (r >= N) r = N - 1;
      bf16_t* xr = Xs + row * 264;
      if (seg < 4) {
        const float4* hp = (const float4*)(h + (size_t)r * DHID + seg * 32);
#pragma unroll
        for (int q = 0; q < 8; ++q) {
          float4 v = hp[q];
          bf16_t* o = xr + seg * 32 + q * 4;
          o[0] = f2bf(v.x); o[1] = f2bf(v.y); o[2] = f2bf(v.z); o[3] = f2bf(v.w);
        }
      } else {
        const float sc = inv_d[r];
        const float4* sp = (const float4*)(S + (size_t)r * DHID + (seg - 4) * 32);
#pragma unroll
        for (int q = 0; q < 8; ++q) {
          float4 v = sp[q];
          bf16_t* o = xr + 128 + (seg - 4) * 32 + q * 4;
          o[0] = f2bf(sc * v.x); o[1] = f2bf(sc * v.y);
          o[2] = f2bf(sc * v.z); o[3] = f2bf(sc * v.w);
        }
      }
    }
    __syncthreads();

    v8f acc00 = {}, acc01 = {}, acc10 = {}, acc11 = {};
#pragma unroll
    for (int k = 0; k < 8; ++k) {
      const int kb = k * 32 + half * 16;   // A layout: 16 contiguous K per half-wave
      union { v16bf v; int4 q[2]; } a0, a1;
      const bf16_t* ap0 = Xs + (size_t)lm * 264 + kb;
      const bf16_t* ap1 = Xs + (size_t)(16 + lm) * 264 + kb;
      a0.q[0] = *(const int4*)(ap0); a0.q[1] = *(const int4*)(ap0 + 8);
      a1.q[0] = *(const int4*)(ap1); a1.q[1] = *(const int4*)(ap1 + 8);
      acc00 = __builtin_amdgcn_wmma_f32_16x16x32_bf16(false, a0.v, false, Bf[0][k], (short)0, acc00, false, false);
      acc10 = __builtin_amdgcn_wmma_f32_16x16x32_bf16(false, a1.v, false, Bf[0][k], (short)0, acc10, false, false);
      acc01 = __builtin_amdgcn_wmma_f32_16x16x32_bf16(false, a0.v, false, Bf[1][k], (short)0, acc01, false, false);
      acc11 = __builtin_amdgcn_wmma_f32_16x16x32_bf16(false, a1.v, false, Bf[1][k], (short)0, acc11, false, false);
    }
    __syncthreads();            // done reading Xs; alias smem as As

    // C layout: VGPR j -> row j+8*half, col lm
#pragma unroll
    for (int j = 0; j < 8; ++j) {
      const int cc = wv * 32 + lm;
      As[(j + half * 8)      * 260 + cc     ] = acc00[j];
      As[(16 + j + half * 8) * 260 + cc     ] = acc10[j];
      As[(j + half * 8)      * 260 + cc + 16] = acc01[j];
      As[(16 + j + half * 8) * 260 + cc + 16] = acc11[j];
    }
    __syncthreads();

    {                           // epilogue: 16 contiguous cols per thread
      const int row = t >> 3;
      const int cb  = (t & 7) * 16;
      const int r   = r0 + row;
      if (r < N) {
        const float id2 = inv_d2[r];
        const float idv = inv_d[r];
        const float* ar = As + row * 260;
        const float4* hp = (const float4*)(h  + (size_t)r * DHID + cb);
        const float4* sp = (const float4*)(S  + (size_t)r * DHID + cb);
        const float4* rp = (const float4*)(Rl + cb);
        float4* op = (float4*)(h_next + (size_t)r * DHID + cb);
        float4* ep = (float4*)(emb    + (size_t)r * DHID + cb);
#pragma unroll
        for (int q = 0; q < 4; ++q) {
          float4 hv = hp[q], sv = sp[q], rv = rp[q], ev = ep[q];
          float hh[4] = {hv.x, hv.y, hv.z, hv.w};
          float ss[4] = {sv.x, sv.y, sv.z, sv.w};
          float rr[4] = {rv.x, rv.y, rv.z, rv.w};
          float oo[4];
#pragma unroll
          for (int j = 0; j < 4; ++j) {
            const int c = cb + q * 4 + j;
            float g = ar[c];
            float b = ar[128 + c];
            g = (g > 0.f ? g : 0.2f * g) + 1.0f;
            b = (b > 0.f ? b : 0.2f * b);
            const float nb = idv * ss[j];
            const float m  = hh[j] + g * rr[j] + b - nb;
            oo[j] = id2 * (ss[j] + hh[j] + m);
          }
          float4 o4; o4.x = oo[0]; o4.y = oo[1]; o4.z = oo[2]; o4.w = oo[3];
          op[q] = o4;
          float4 e4; e4.x = ev.x + oo[0]; e4.y = ev.y + oo[1];
          e4.z = ev.z + oo[2]; e4.w = ev.w + oo[3];
          ep[q] = e4;
        }
      }
    }
  }
}

__global__ void k_combine(const int* __restrict__ deg, const float* __restrict__ accT,
                          float* __restrict__ out, int total) {
  int i = blockIdx.x * blockDim.x + threadIdx.x;
  if (i < total) {
    const int row = i >> 7;                  // DHID == 128
    const float v = (deg[row] > TAILK) ? out[i] : accT[i];
    out[i] = 0.25f * v;                      // mean over 4 emb states
  }
}

// -------------------------------------------------------------------- host
extern "C" void kernel_launch(void* const* d_in, const int* in_sizes, int n_in,
                              void* d_out, int out_size, void* d_ws, size_t ws_size,
                              hipStream_t stream) {
  const float* x  = (const float*)d_in[0];
  const float* G1 = (const float*)d_in[1];
  const float* G2 = (const float*)d_in[2];
  const float* B1 = (const float*)d_in[3];
  const float* B2 = (const float*)d_in[4];
  const float* R  = (const float*)d_in[5];
  const int* esrc = (const int*)d_in[6];
  const int* edst = (const int*)d_in[7];
  const int N = in_sizes[0] / DHID;
  const int E = in_sizes[6];
  const int L = in_sizes[5] / DHID;
  float* out = (float*)d_out;

  char* p = (char*)d_ws;
  auto alloc = [&](size_t bytes) -> void* {
    void* r = (void*)p; p += (bytes + 255) & ~(size_t)255; return r;
  };
  const int NB = (N + 255) / 256;
  int*    deg    = (int*)alloc((size_t)N * 4);
  int*    rs     = (int*)alloc((size_t)(N + 1) * 4);
  int*    bsum   = (int*)alloc((size_t)NB * 4);
  int*    rc     = (int*)alloc((size_t)N * 4);
  int*    csr    = (int*)alloc((size_t)E * 4);
  float*  inv_d  = (float*)alloc((size_t)N * 4);
  float*  inv_d1 = (float*)alloc((size_t)N * 4);
  float*  inv_d2 = (float*)alloc((size_t)N * 4);
  bf16_t* Wt     = (bf16_t*)alloc(256 * 256 * 2);
  float*  hA     = (float*)alloc((size_t)N * DHID * 4);
  float*  hB     = (float*)alloc((size_t)N * DHID * 4);
  float*  Sbuf   = (float*)alloc((size_t)N * DHID * 4);
  float*  accT   = (float*)alloc((size_t)N * DHID * 4);

  const int total = N * DHID;
  const int gT = (total + 255) / 256;
  const int gE = (E + 255) / 256;
  const int rowsumGrid = (N + 7) / 8;        // 8 waves/block, wave per row
  const int nMB = (N + 31) / 32;
  const int gemmGrid = nMB < 1024 ? nMB : 1024;

  // ---- graph preprocessing: degrees -> CSR
  k_zero_i32<<<NB, 256, 0, stream>>>(deg, N);
  k_deg<<<gE, 256, 0, stream>>>(esrc, deg, E);
  k_scan_block<<<NB, 256, 0, stream>>>(deg, rs, bsum, N);
  k_scan_bsums<<<1, 1024, 0, stream>>>(bsum, NB);
  k_scan_add<<<NB, 256, 0, stream>>>(rs, bsum, deg, rc, N, E);
  k_node_params<<<NB, 256, 0, stream>>>(deg, inv_d, inv_d1, inv_d2, N);
  k_csr_fill<<<gE, 256, 0, stream>>>(esrc, edst, rc, csr, E);

  // ---- head pass: pure SpMM chain, emb accumulates in d_out (seeded with x)
  k_init<<<gT, 256, 0, stream>>>(x, hA, out, total);
  float* hc = hA; float* hn = hB;
  for (int l = 0; l < L; ++l) {
    k_rowsum<<<rowsumGrid, 256, 0, stream>>>(hc, rs, csr, hn, out, inv_d1, N, 1);
    float* tmp = hc; hc = hn; hn = tmp;
  }

  // ---- tail pass: SpMM + bf16 WMMA gamma/beta + fused update
  k_init<<<gT, 256, 0, stream>>>(x, hA, accT, total);
  hc = hA; hn = hB;
  for (int l = 0; l < L; ++l) {
    k_rowsum<<<rowsumGrid, 256, 0, stream>>>(hc, rs, csr, Sbuf, (float*)0, inv_d1, N, 0);
    k_pack_w<<<256, 256, 0, stream>>>(G1, G2, B1, B2, l, Wt);
    k_tail_gemm<<<gemmGrid, 256, 0, stream>>>(hc, Sbuf, Wt, R + (size_t)l * DHID,
                                              inv_d, inv_d2, hn, accT, N, nMB);
    float* tmp = hc; hc = hn; hn = tmp;
  }

  // ---- select head/tail per node, apply 1/4 mean
  k_combine<<<gT, 256, 0, stream>>>(deg, accT, out, total);
}
